// GraphSAGELayer_22187801051297
// MI455X (gfx1250) — matrix operand order repaired and checked by reading, not compiled, not run on hardware
//
#include <hip/hip_runtime.h>
#include <hip/hip_bf16.h>

// ---------- types ----------
typedef __attribute__((ext_vector_type(16))) __bf16        v16bf;
typedef __attribute__((ext_vector_type(8)))  float         v8f;
typedef __attribute__((ext_vector_type(4)))  float         f32x4;
typedef __attribute__((ext_vector_type(4)))  unsigned int  u32x4;

#define B_      8
#define N_      4096
#define K_      32
#define H_      256
#define KDIM    512          // concatenated input dim: [h | h_agg]
#define MTILE   64           // rows per workgroup
#define APITCH  520          // bf16 elems per LDS A-row (512 + 8 pad -> 1040B, 16B aligned)
#define OPITCH  260          // f32 elems per LDS out-row (1040B, 16B aligned)

// ---------- bf16 helpers (round-to-nearest-even) ----------
static __device__ __forceinline__ unsigned short bf16_hi(float f) {
  unsigned u = __float_as_uint(f);
  u += 0x7FFFu + ((u >> 16) & 1u);
  return (unsigned short)(u >> 16);
}
static __device__ __forceinline__ float bf16_to_f(unsigned short h) {
  return __uint_as_float(((unsigned)h) << 16);
}

// split 8 floats into packed bf16 hi/lo and store 16B each
static __device__ __forceinline__ void split_store8(unsigned short* hi_p,
                                                    unsigned short* lo_p,
                                                    f32x4 a, f32x4 b) {
  float vals[8] = {a.x, a.y, a.z, a.w, b.x, b.y, b.z, b.w};
  unsigned int hh[4], ll[4];
#pragma unroll
  for (int i = 0; i < 4; ++i) {
    unsigned short h0 = bf16_hi(vals[2 * i]);
    unsigned short h1 = bf16_hi(vals[2 * i + 1]);
    float r0 = vals[2 * i]     - bf16_to_f(h0);
    float r1 = vals[2 * i + 1] - bf16_to_f(h1);
    hh[i] = (unsigned)h0 | ((unsigned)h1 << 16);
    ll[i] = (unsigned)bf16_hi(r0) | ((unsigned)bf16_hi(r1) << 16);
  }
  u32x4 qh = {hh[0], hh[1], hh[2], hh[3]};
  u32x4 ql = {ll[0], ll[1], ll[2], ll[3]};
  *(u32x4*)hi_p = qh;
  *(u32x4*)lo_p = ql;
}

// ---------- kernel 1: split [W_self ; W_neigh] into bf16 hi/lo Wcat[256][512] ----------
__global__ void sage_wprep_kernel(const float* __restrict__ Wself,
                                  const float* __restrict__ Wneigh,
                                  unsigned short* __restrict__ Whi,
                                  unsigned short* __restrict__ Wlo) {
  int gid = blockIdx.x * blockDim.x + threadIdx.x;  // 0 .. 256*512-1
  int n = gid >> 9;
  int k = gid & 511;
  float v = (k < H_) ? Wself[n * H_ + k] : Wneigh[n * H_ + (k - H_)];
  unsigned short hi = bf16_hi(v);
  Whi[gid] = hi;
  Wlo[gid] = bf16_hi(v - bf16_to_f(hi));
}

// ---------- kernel 2: fused gather+aggregate -> dual GEMM -> GELU -> LayerNorm ----------
__global__ void __launch_bounds__(256)
sage_main_kernel(const float* __restrict__ h,
                 const long long* __restrict__ nbr_idx,
                 const float* __restrict__ nbr_w,
                 const unsigned short* __restrict__ Whi,
                 const unsigned short* __restrict__ Wlo,
                 const float* __restrict__ gamma,
                 const float* __restrict__ beta,
                 float* __restrict__ out) {
  extern __shared__ char smem[];
  unsigned short* a_hi = (unsigned short*)smem;             // [MTILE][APITCH]
  unsigned short* a_lo = a_hi + MTILE * APITCH;             // [MTILE][APITCH]

  const int lane  = threadIdx.x & 31;
  const int wave  = threadIdx.x >> 5;
  const int bidx  = blockIdx.x >> 6;                        // N_/MTILE = 64 tiles/batch
  const int node0 = (blockIdx.x & 63) * MTILE;

  const float* hb = h + (long long)bidx * N_ * H_;

  // ================= Phase A: gather + weighted aggregate, stage split-bf16 A =================
  for (int rr = 0; rr < 8; ++rr) {
    const int row  = wave * 8 + rr;                          // uniform per wave
    const int node = node0 + row;
    const long long* ip = nbr_idx + ((long long)bidx * N_ + node) * K_;
    const float*     wp = nbr_w   + ((long long)bidx * N_ + node) * K_;

    f32x4 acc0 = {}, acc1 = {};
#pragma unroll 4
    for (int j = 0; j < K_; ++j) {
      const long long nid = ip[j];                           // wave-uniform
      const float     wj  = wp[j];
      const f32x4* hp = (const f32x4*)(hb + nid * (long long)H_ + lane * 8);
      acc0 += wj * hp[0];
      acc1 += wj * hp[1];
    }
    // aggregate features -> cols [256 + lane*8)
    split_store8(a_hi + row * APITCH + 256 + lane * 8,
                 a_lo + row * APITCH + 256 + lane * 8, acc0, acc1);
    // self features -> cols [lane*8)
    const f32x4* hs = (const f32x4*)(hb + (long long)node * H_ + lane * 8);
    split_store8(a_hi + row * APITCH + lane * 8,
                 a_lo + row * APITCH + lane * 8, hs[0], hs[1]);
  }
  __syncthreads();

  // ================= GEMM: [64 x 512]bf16x2 @ Wcat^T -> [64 x 256]f32 =================
  const int wm = wave & 3;                    // which 16-row tile
  const int wn = wave >> 2;                   // which 128-col half
  v8f accv[8];
#pragma unroll
  for (int t = 0; t < 8; ++t) accv[t] = (v8f){};

  const int arow  = wm * 16 + (lane & 15);
  const int aoff0 = (lane >> 4) * 16;         // A frag: K+{0,16}
  const int bK    = (lane >> 4) * 16;         // B frag: K+{0,16}
  const int bN0   = wn * 128 + (lane & 15);

  union Frag { u32x4 q[2]; v16bf v; };

  for (int kt = 0; kt < KDIM / 32; ++kt) {
    Frag fah, fal;
    {
      const unsigned short* ap = a_hi + arow * APITCH + kt * 32 + aoff0;
      fah.q[0] = *(const u32x4*)(ap);
      fah.q[1] = *(const u32x4*)(ap + 8);      // contiguous 32B
      const unsigned short* alp = a_lo + arow * APITCH + kt * 32 + aoff0;
      fal.q[0] = *(const u32x4*)(alp);
      fal.q[1] = *(const u32x4*)(alp + 8);
    }
#pragma unroll
    for (int nt = 0; nt < 8; ++nt) {
      Frag fbh, fbl;
      const unsigned short* bp = Whi + (size_t)(bN0 + nt * 16) * KDIM + kt * 32 + bK;
      fbh.q[0] = *(const u32x4*)(bp);
      fbh.q[1] = *(const u32x4*)(bp + 8);      // contiguous 32B
      const unsigned short* blp = Wlo + (size_t)(bN0 + nt * 16) * KDIM + kt * 32 + bK;
      fbl.q[0] = *(const u32x4*)(blp);
      fbl.q[1] = *(const u32x4*)(blp + 8);

      // split-precision: hi*hi + hi*lo + lo*hi (drop lo*lo)
      accv[nt] = __builtin_amdgcn_wmma_f32_16x16x32_bf16(
          false, fah.v, false, fbh.v, (short)0, accv[nt], false, false);
      accv[nt] = __builtin_amdgcn_wmma_f32_16x16x32_bf16(
          false, fah.v, false, fbl.v, (short)0, accv[nt], false, false);
      accv[nt] = __builtin_amdgcn_wmma_f32_16x16x32_bf16(
          false, fal.v, false, fbh.v, (short)0, accv[nt], false, false);
    }
  }

  // ================= Epilogue: GELU(exact erf) -> LDS, LayerNorm -> out =================
  __syncthreads();                            // A-tile dead; alias smem as f32 out-tile
  float* outT  = (float*)smem;                // [MTILE][OPITCH]
  float* psum  = (float*)(smem + MTILE * OPITCH * 4);
  float* psq   = psum + 256;
  float* pmean = psq + 256;
  float* prstd = pmean + 64;

#pragma unroll
  for (int nt = 0; nt < 8; ++nt) {
    const int c = wn * 128 + nt * 16 + (lane & 15);
#pragma unroll
    for (int v = 0; v < 8; ++v) {
      const int m = wm * 16 + v + ((lane >> 4) << 3);
      float x = accv[nt][v];
      float g = 0.5f * x * (1.0f + erff(x * 0.70710678118654752f));
      outT[m * OPITCH + c] = g;
    }
  }
  __syncthreads();

  {
    const int row = threadIdx.x >> 2;
    const int seg = threadIdx.x & 3;
    const float* rp = outT + row * OPITCH + seg * 64;
    float s = 0.f, sq = 0.f;
#pragma unroll 8
    for (int i = 0; i < 64; ++i) { float x = rp[i]; s += x; sq += x * x; }
    psum[row * 4 + seg] = s;
    psq[row * 4 + seg]  = sq;
  }
  __syncthreads();

  if (threadIdx.x < 64) {
    const int row = threadIdx.x;
    float s  = psum[row * 4] + psum[row * 4 + 1] + psum[row * 4 + 2] + psum[row * 4 + 3];
    float sq = psq[row * 4]  + psq[row * 4 + 1]  + psq[row * 4 + 2]  + psq[row * 4 + 3];
    float mean = s * (1.0f / 256.0f);
    float var  = sq * (1.0f / 256.0f) - mean * mean;
    pmean[row] = mean;
    prstd[row] = rsqrtf(var + 1e-5f);
  }
  __syncthreads();

  {
    const int row = threadIdx.x >> 2;
    const int seg = threadIdx.x & 3;
    const float mean = pmean[row];
    const float rstd = prstd[row];
    const float* rp = outT + row * OPITCH + seg * 64;
    float* op = out + ((long long)bidx * N_ + node0 + row) * H_ + seg * 64;
#pragma unroll
    for (int i = 0; i < 16; ++i) {
      f32x4 x = *(const f32x4*)(rp + i * 4);
      const int c = seg * 64 + i * 4;
      f32x4 y;
      y.x = (x.x - mean) * rstd * gamma[c + 0] + beta[c + 0];
      y.y = (x.y - mean) * rstd * gamma[c + 1] + beta[c + 1];
      y.z = (x.z - mean) * rstd * gamma[c + 2] + beta[c + 2];
      y.w = (x.w - mean) * rstd * gamma[c + 3] + beta[c + 3];
      *(f32x4*)(op + i * 4) = y;
    }
  }
}

extern "C" void kernel_launch(void* const* d_in, const int* in_sizes, int n_in,
                              void* d_out, int out_size, void* d_ws, size_t ws_size,
                              hipStream_t stream) {
  const float*     h_ptr   = (const float*)d_in[0];
  const long long* nbr_idx = (const long long*)d_in[1];   // reference dtype: int64
  const float*     nbr_w   = (const float*)d_in[2];
  const float*     Wself   = (const float*)d_in[3];
  const float*     Wneigh  = (const float*)d_in[4];
  const float*     gamma   = (const float*)d_in[5];
  const float*     beta    = (const float*)d_in[6];
  float*           out     = (float*)d_out;

  unsigned short* Whi = (unsigned short*)d_ws;            // 256*512 bf16 = 256KB
  unsigned short* Wlo = Whi + H_ * KDIM;                  // next 256KB

  sage_wprep_kernel<<<(H_ * KDIM) / 256, 256, 0, stream>>>(Wself, Wneigh, Whi, Wlo);

  const size_t lds_bytes = (size_t)2 * MTILE * APITCH * sizeof(unsigned short); // 133120B
  (void)hipFuncSetAttribute((const void*)sage_main_kernel,
                            hipFuncAttributeMaxDynamicSharedMemorySize,
                            (int)lds_bytes);
  sage_main_kernel<<<dim3(B_ * (N_ / MTILE)), dim3(256), lds_bytes, stream>>>(
      h_ptr, nbr_idx, nbr_w, Whi, Wlo, gamma, beta, out);
}
